// UniversalGNN_38766374814296
// MI455X (gfx1250) — matrix operand order repaired and checked by reading, not compiled, not run on hardware
//
#include <hip/hip_runtime.h>
#include <hip/hip_bf16.h>
#include <stdint.h>

// ---------------------------------------------------------------------------
// CDNA5 (gfx1250) WMMA types / helpers
// ---------------------------------------------------------------------------
typedef __attribute__((ext_vector_type(16))) __bf16 v16bf;
typedef __attribute__((ext_vector_type(8)))  __bf16 v8bf;
typedef __attribute__((ext_vector_type(8)))  float  v8f;

__device__ __forceinline__ float silu_f(float x) {
    return x / (1.0f + __expf(-x));
}

// ---------------------------------------------------------------------------
// gfx1250 async global->LDS copy (ASYNCcnt-tracked, bypasses VGPRs).
// LDS operand is the workgroup-relative byte offset = low 32 bits of the
// generic pointer (AMDGPU local->flat addrspacecast keeps offset in [31:0]).
// ---------------------------------------------------------------------------
__device__ __forceinline__ void async_copy_b128(uint32_t lds_off, const void* gaddr) {
    unsigned long long ga = (unsigned long long)(uintptr_t)gaddr;
    asm volatile("global_load_async_to_lds_b128 %0, %1, off"
                 :: "v"(lds_off), "v"(ga)
                 : "memory");
}
__device__ __forceinline__ void wait_async_zero() {
    asm volatile("s_wait_asynccnt 0x0" ::: "memory");
}

// A fragment (16x32 bf16, MxK). Per CDNA5 ISA 16-bit A layout:
// lane l: row m = l&15; elements 0..7 hold K = kbase + 8*(l>>4) + 0..7,
// elements 8..15 hold K = kbase + 16 + 8*(l>>4) + 0..7.
// Source row-major bf16, row stride 64 -> two contiguous 16B LDS loads.
__device__ __forceinline__ v16bf load_fragA(const __bf16* base, int row, int kbase, int lane) {
    int hl = (lane >> 4) & 1;
    const __bf16* p0 = base + row * 64 + kbase + 8 * hl;
    const __bf16* p1 = p0 + 16;
    v8bf lo = *(const v8bf*)p0;
    v8bf hi = *(const v8bf*)p1;
    v16bf r;
#pragma unroll
    for (int i = 0; i < 8; ++i) { r[i] = lo[i]; r[8 + i] = hi[i]; }
    return r;
}

// Same A fragment but sourced from an f32 row-major LDS tile (convert on load).
__device__ __forceinline__ v16bf load_fragA_f32(const float* base, int row, int kbase, int lane) {
    int hl = (lane >> 4) & 1;
    const float* p0 = base + row * 64 + kbase + 8 * hl;
    const float* p1 = p0 + 16;
    v8f lo = *(const v8f*)p0;
    v8f hi = *(const v8f*)p1;
    v16bf r;
#pragma unroll
    for (int i = 0; i < 8; ++i) { r[i] = (__bf16)lo[i]; r[8 + i] = (__bf16)hi[i]; }
    return r;
}

// B fragment (32x16 bf16, KxN). lane l: col n = l&15;
// elements 0..7 hold K = kbase + 8*(l>>4) + 0..7, elements 8..15 hold K = kbase + 16 + 8*(l>>4) + 0..7.
// Weights staged transposed in LDS: WT[n*64 + k] -> two 16B contiguous loads.
__device__ __forceinline__ v16bf load_fragB(const __bf16* wt, int n, int kbase, int lane) {
    int hl = (lane >> 4) & 1;
    const __bf16* p0 = wt + n * 64 + kbase + 8 * hl;
    const __bf16* p1 = p0 + 16;
    v8bf lo = *(const v8bf*)p0;
    v8bf hi = *(const v8bf*)p1;
    v16bf r;
#pragma unroll
    for (int i = 0; i < 8; ++i) { r[i] = lo[i]; r[8 + i] = hi[i]; }
    return r;
}

// One wave computes a 16-row x 64-col tile: D = A(16x64) * W(64x64), f32 acc.
__device__ __forceinline__ void gemm_tile_16x64(const __bf16* A, const __bf16* WT,
                                                int lane, v8f acc[4]) {
#pragma unroll
    for (int kc = 0; kc < 2; ++kc) {
        v16bf a = load_fragA(A, lane & 15, kc * 32, lane);
#pragma unroll
        for (int nt = 0; nt < 4; ++nt) {
            v16bf b = load_fragB(WT, nt * 16 + (lane & 15), kc * 32, lane);
            acc[nt] = __builtin_amdgcn_wmma_f32_16x16x32_bf16(
                false, a, false, b, (short)0, acc[nt], false, false);
        }
    }
}

__device__ __forceinline__ void gemm_tile_16x64_f32A(const float* A, const __bf16* WT,
                                                     int lane, v8f acc[4]) {
#pragma unroll
    for (int kc = 0; kc < 2; ++kc) {
        v16bf a = load_fragA_f32(A, lane & 15, kc * 32, lane);
#pragma unroll
        for (int nt = 0; nt < 4; ++nt) {
            v16bf b = load_fragB(WT, nt * 16 + (lane & 15), kc * 32, lane);
            acc[nt] = __builtin_amdgcn_wmma_f32_16x16x32_bf16(
                false, a, false, b, (short)0, acc[nt], false, false);
        }
    }
}

// Stage a 64x64 f32 weight matrix into LDS as transposed bf16 (WT[n*64+k]).
__device__ __forceinline__ void stage_wt(const float* __restrict__ W, __bf16* wt, int t) {
    for (int i = t; i < 64 * 64; i += 256) {
        int k = i >> 6, n = i & 63;
        wt[n * 64 + k] = (__bf16)W[i];
    }
}

// ---------------------------------------------------------------------------
// Kernel: t_feat[g][c] = silu(time_emb[g]) @ Wt + bt   (64x32 @ 32x64, tiny)
// ---------------------------------------------------------------------------
__global__ __launch_bounds__(256) void k_timefeat(const float* __restrict__ time_emb,
                                                  const float* __restrict__ Wt,
                                                  const float* __restrict__ bt,
                                                  float* __restrict__ tfeat) {
    int i = blockIdx.x * 256 + threadIdx.x;
    if (i >= 64 * 64) return;
    int g = i >> 6, c = i & 63;
    float s = bt[c];
#pragma unroll 8
    for (int k = 0; k < 32; ++k) {
        s += silu_f(time_emb[g * 32 + k]) * Wt[k * 64 + c];
    }
    tfeat[i] = s;
}

// ---------------------------------------------------------------------------
// Kernel: zero-fill (agg buffer), float4 stores
// ---------------------------------------------------------------------------
__global__ __launch_bounds__(256) void k_zero(float4* __restrict__ p, long long n4) {
    long long i = (long long)blockIdx.x * 256 + threadIdx.x;
    if (i < n4) p[i] = make_float4(0.f, 0.f, 0.f, 0.f);
}

// ---------------------------------------------------------------------------
// Kernel: node init. h = (silu(x@W1+b1))@W2 + b2 + tfeat[batch]
//   x tile async-staged to LDS; layer1 K=7 on VALU -> bf16 tile; layer2 WMMA.
// ---------------------------------------------------------------------------
__global__ __launch_bounds__(256) void k_node_init(
    const float* __restrict__ x, const int* __restrict__ batch,
    const float* __restrict__ W1, const float* __restrict__ b1,
    const float* __restrict__ W2, const float* __restrict__ b2,
    const float* __restrict__ tfeat, float* __restrict__ h, int nrows) {
    __shared__ __attribute__((aligned(32))) __bf16 sTile[128 * 64];
    __shared__ __attribute__((aligned(32))) __bf16 sWT[64 * 64];
    __shared__ __attribute__((aligned(16))) float  sX[128 * 7];
    int t = threadIdx.x;
    int rowBase = blockIdx.x * 128;

    stage_wt(W2, sWT, t);

    // stage x tile (128*7 f32 = 224 x 16B, contiguous)
    bool full = (rowBase + 128 <= nrows);
    if (full) {
        if (t < 224) {  // waves 0..6 fully active
            async_copy_b128((uint32_t)(uintptr_t)(sX + t * 4),
                            (const void*)(x + (size_t)rowBase * 7 + t * 4));
        }
        wait_async_zero();
    } else {
        for (int i = t; i < 128 * 7; i += 256) {
            int gi = rowBase * 7 + i;
            sX[i] = (gi < nrows * 7) ? x[gi] : 0.f;
        }
    }
    __syncthreads();

    // layer 1: thread handles (row, 32 cols)
    int lr = t >> 1;
    int c0 = (t & 1) * 32;
    float xv[7];
#pragma unroll
    for (int k = 0; k < 7; ++k) xv[k] = sX[lr * 7 + k];
    for (int c = c0; c < c0 + 32; ++c) {
        float s = b1[c];
#pragma unroll
        for (int k = 0; k < 7; ++k) s += xv[k] * W1[k * 64 + c];
        sTile[lr * 64 + c] = (__bf16)silu_f(s);
    }
    __syncthreads();

    // layer 2: WMMA
    int wave = t >> 5, lane = t & 31;
    const __bf16* A = sTile + wave * 16 * 64;
    v8f acc[4] = {};
    gemm_tile_16x64(A, sWT, lane, acc);

    int hl = (lane >> 4) & 1, n0 = lane & 15;
#pragma unroll
    for (int r = 0; r < 8; ++r) {
        int grow = rowBase + wave * 16 + r + 8 * hl;
        if (grow < nrows) {
            int bg = batch[grow];
#pragma unroll
            for (int nt = 0; nt < 4; ++nt) {
                int col = nt * 16 + n0;
                h[grow * 64 + col] = acc[nt][r] + b2[col] + tfeat[bg * 64 + col];
            }
        }
    }
}

// ---------------------------------------------------------------------------
// Kernel: fused edge MLP + message + scatter for one GINE conv.
//   ef   = (silu(edge_attr@W1e+b1e))@W2e + b2e       (layer2 via WMMA)
//   msg  = relu(h[src] + ef)
//   agg[dst] += msg                                  (f32 hardware atomics)
// edge_feat recomputed each conv (307MB > L2 if materialized; WMMA is free)
// ---------------------------------------------------------------------------
__global__ __launch_bounds__(256) void k_scatter(
    const float* __restrict__ edge_attr,
    const int* __restrict__ src, const int* __restrict__ dst,
    const float* __restrict__ W1, const float* __restrict__ b1,
    const float* __restrict__ W2, const float* __restrict__ b2,
    const float* __restrict__ h, float* __restrict__ agg, int nedges) {
    __shared__ __attribute__((aligned(32))) __bf16 sTile[128 * 64];
    __shared__ __attribute__((aligned(32))) __bf16 sWT[64 * 64];
    __shared__ __attribute__((aligned(16))) float  sEA[128 * 2];
    int t = threadIdx.x;
    int eBase = blockIdx.x * 128;

    stage_wt(W2, sWT, t);

    // stage edge_attr block (256 f32 = 64 x 16B, contiguous)
    bool full = (eBase + 128 <= nedges);
    if (full) {
        if (t < 64) {  // waves 0..1 fully active
            async_copy_b128((uint32_t)(uintptr_t)(sEA + t * 4),
                            (const void*)(edge_attr + (size_t)eBase * 2 + t * 4));
        }
        wait_async_zero();
    } else {
        int gi = eBase * 2 + t;
        sEA[t] = (gi < nedges * 2) ? edge_attr[gi] : 0.f;
    }
    __syncthreads();

    // edge layer 1 (K=2) on VALU
    int lr = t >> 1;
    float a0 = sEA[lr * 2 + 0];
    float a1 = sEA[lr * 2 + 1];
    int c0 = (t & 1) * 32;
    for (int c = c0; c < c0 + 32; ++c) {
        float s = b1[c] + a0 * W1[c] + a1 * W1[64 + c];
        sTile[lr * 64 + c] = (__bf16)silu_f(s);
    }
    __syncthreads();

    // edge layer 2: WMMA
    int wave = t >> 5, lane = t & 31;
    const __bf16* A = sTile + wave * 16 * 64;
    v8f acc[4] = {};
    gemm_tile_16x64(A, sWT, lane, acc);

    // message + scatter
    int hl = (lane >> 4) & 1, n0 = lane & 15;
#pragma unroll
    for (int r = 0; r < 8; ++r) {
        int ee = eBase + wave * 16 + r + 8 * hl;
        if (ee < nedges) {
            int sn = src[ee];
            int dn = dst[ee];
#pragma unroll
            for (int nt = 0; nt < 4; ++nt) {
                int col = nt * 16 + n0;
                float msg = acc[nt][r] + b2[col] + h[(long)sn * 64 + col];
                msg = fmaxf(msg, 0.f);
                __hip_atomic_fetch_add(&agg[(long)dn * 64 + col], msg,
                                       __ATOMIC_RELAXED, __HIP_MEMORY_SCOPE_AGENT);
            }
        }
    }
}

// ---------------------------------------------------------------------------
// Kernel: node update for one conv.
//   u = h + agg ; hout = silu( (silu(u@W1+b1))@W2 + b2 )   (both layers WMMA)
// ---------------------------------------------------------------------------
__global__ __launch_bounds__(256) void k_update(
    const float* __restrict__ hin, const float* __restrict__ agg,
    const float* __restrict__ W1, const float* __restrict__ b1,
    const float* __restrict__ W2, const float* __restrict__ b2,
    float* __restrict__ hout, int nrows, int do_silu_out) {
    __shared__ __attribute__((aligned(32))) __bf16 sTile[128 * 64];
    __shared__ __attribute__((aligned(32))) __bf16 sWT1[64 * 64];
    __shared__ __attribute__((aligned(32))) __bf16 sWT2[64 * 64];
    int t = threadIdx.x;
    int rowBase = blockIdx.x * 128;

    stage_wt(W1, sWT1, t);
    stage_wt(W2, sWT2, t);

    // stage u = hin + agg as bf16 (coalesced)
    for (int i = t; i < 128 * 64; i += 256) {
        int r = i >> 6, c = i & 63;
        int grow = rowBase + r;
        float u = 0.f;
        if (grow < nrows) u = hin[grow * 64 + c] + agg[grow * 64 + c];
        sTile[i] = (__bf16)u;
    }
    __syncthreads();

    int wave = t >> 5, lane = t & 31;
    __bf16* A = sTile + wave * 16 * 64;

    // layer 1 (WMMA)
    v8f acc[4] = {};
    gemm_tile_16x64(A, sWT1, lane, acc);

    // tmp = silu(acc + b1) -> back into own LDS strip
    int hl = (lane >> 4) & 1, n0 = lane & 15;
#pragma unroll
    for (int nt = 0; nt < 4; ++nt) {
        int col = nt * 16 + n0;
        float bb = b1[col];
#pragma unroll
        for (int r = 0; r < 8; ++r) {
            A[(r + 8 * hl) * 64 + col] = (__bf16)silu_f(acc[nt][r] + bb);
        }
    }
    __syncthreads();

    // layer 2 (WMMA)
    v8f acc2[4] = {};
    gemm_tile_16x64(A, sWT2, lane, acc2);

#pragma unroll
    for (int r = 0; r < 8; ++r) {
        int grow = rowBase + wave * 16 + r + 8 * hl;
        if (grow < nrows) {
#pragma unroll
            for (int nt = 0; nt < 4; ++nt) {
                int col = nt * 16 + n0;
                float s = acc2[nt][r] + b2[col];
                if (do_silu_out) s = silu_f(s);
                hout[grow * 64 + col] = s;
            }
        }
    }
}

// ---------------------------------------------------------------------------
// Kernel: final MLP. out = (silu(h@W1+b1)) @ W2(64x1) + b2
//   h tile async-staged to LDS as f32 (32KB); bf16 convert at fragment load;
//   SiLU temp kept f32 in LDS for the final 64->1 dot.
// ---------------------------------------------------------------------------
__global__ __launch_bounds__(256) void k_final(
    const float* __restrict__ hin,
    const float* __restrict__ W1, const float* __restrict__ b1,
    const float* __restrict__ W2, const float* __restrict__ b2,
    float* __restrict__ out, int nrows) {
    __shared__ __attribute__((aligned(32))) float  sF[128 * 64];
    __shared__ __attribute__((aligned(32))) __bf16 sWT1[64 * 64];
    int t = threadIdx.x;
    int rowBase = blockIdx.x * 128;

    stage_wt(W1, sWT1, t);

    bool full = (rowBase + 128 <= nrows);
    if (full) {
        const float* g = hin + (size_t)rowBase * 64;
#pragma unroll
        for (int i = 0; i < 8; ++i) {
            int idx = (t + i * 256) * 4;  // chunk start in floats
            async_copy_b128((uint32_t)(uintptr_t)(sF + idx), (const void*)(g + idx));
        }
        wait_async_zero();
    } else {
        for (int i = t; i < 128 * 64; i += 256) {
            int r = i >> 6;
            sF[i] = ((rowBase + r) < nrows) ? hin[(size_t)rowBase * 64 + i] : 0.f;
        }
    }
    __syncthreads();

    int wave = t >> 5, lane = t & 31;
    float* A = sF + wave * 16 * 64;
    v8f acc[4] = {};
    gemm_tile_16x64_f32A(A, sWT1, lane, acc);

    int hl = (lane >> 4) & 1, n0 = lane & 15;
#pragma unroll
    for (int nt = 0; nt < 4; ++nt) {
        int col = nt * 16 + n0;
        float bb = b1[col];
#pragma unroll
        for (int r = 0; r < 8; ++r) {
            A[(r + 8 * hl) * 64 + col] = silu_f(acc[nt][r] + bb);
        }
    }
    __syncthreads();

    // layer 2: 64 -> 1 dot product, one thread per row
    if (t < 128) {
        int grow = rowBase + t;
        if (grow < nrows) {
            float s = b2[0];
#pragma unroll 8
            for (int k = 0; k < 64; ++k) s += sF[t * 64 + k] * W2[k];
            out[grow] = s;
        }
    }
}

// ---------------------------------------------------------------------------
// Host launch
// ---------------------------------------------------------------------------
extern "C" void kernel_launch(void* const* d_in, const int* in_sizes, int n_in,
                              void* d_out, int out_size, void* d_ws, size_t ws_size,
                              hipStream_t stream) {
    const float* x         = (const float*)d_in[0];
    const int*   edge_idx  = (const int*)d_in[1];
    const float* edge_attr = (const float*)d_in[2];
    const float* time_emb  = (const float*)d_in[3];
    const int*   batch     = (const int*)d_in[4];

    // params flattened in dict order: node(W1,b1,W2,b2), edge(..), time(W,b),
    // conv1(..), conv2(..), conv3(..), final(..)
    const float* P[31];
    for (int i = 5; i < 31 && i < n_in; ++i) P[i] = (const float*)d_in[i];

    const int N = in_sizes[0] / 7;   // 100000
    const int E = in_sizes[1] / 2;   // 1200000
    const int* src = edge_idx;       // edge_index[0]
    const int* dst = edge_idx + E;   // edge_index[1]

    // workspace carve-up (256B aligned): tfeat | hA | hB | agg  (~77 MB)
    char* w = (char*)d_ws;
    auto carve = [&](size_t bytes) {
        char* p = w;
        w += (bytes + 255) & ~(size_t)255;
        return p;
    };
    float* tfeat = (float*)carve(64 * 64 * sizeof(float));
    float* hA    = (float*)carve((size_t)N * 64 * sizeof(float));
    float* hB    = (float*)carve((size_t)N * 64 * sizeof(float));
    float* agg   = (float*)carve((size_t)N * 64 * sizeof(float));
    (void)ws_size;

    int nodeBlocks = (N + 127) / 128;
    int edgeBlocks = (E + 127) / 128;
    long long n4 = (long long)N * 16;  // N*64 floats as float4
    int zeroBlocks = (int)((n4 + 255) / 256);

    // 1) time features
    k_timefeat<<<16, 256, 0, stream>>>(time_emb, P[13], P[14], tfeat);

    // 2) node embedding + time add
    k_node_init<<<nodeBlocks, 256, 0, stream>>>(x, batch, P[5], P[6], P[7], P[8],
                                                tfeat, hA, N);

    // 3) three GINE convs (edge MLP fused into scatter; recomputed each conv)
    float* hcur = hA;
    float* hnext = hB;
    for (int i = 0; i < 3; ++i) {
        const float* cW1 = P[15 + 4 * i];
        const float* cb1 = P[16 + 4 * i];
        const float* cW2 = P[17 + 4 * i];
        const float* cb2 = P[18 + 4 * i];
        k_zero<<<zeroBlocks, 256, 0, stream>>>((float4*)agg, n4);
        k_scatter<<<edgeBlocks, 256, 0, stream>>>(edge_attr, src, dst,
                                                  P[9], P[10], P[11], P[12],
                                                  hcur, agg, E);
        k_update<<<nodeBlocks, 256, 0, stream>>>(hcur, agg, cW1, cb1, cW2, cb2,
                                                 hnext, N, 1);
        float* tmp = hcur; hcur = hnext; hnext = tmp;
    }

    // 4) final MLP -> d_out [N,1]
    k_final<<<nodeBlocks, 256, 0, stream>>>(hcur, P[27], P[28], P[29], P[30],
                                            (float*)d_out, N);
}
